// MultiScaleQTable_71347996721422
// MI455X (gfx1250) — compile-verified, matrix operand order checked by reading, not run on hardware
//
#include <hip/hip_runtime.h>

// ---------------------------------------------------------------------------
// Types
// ---------------------------------------------------------------------------
typedef __attribute__((ext_vector_type(16))) __bf16 v16bf;
typedef __attribute__((ext_vector_type(8)))  __bf16 v8bf;
typedef __attribute__((ext_vector_type(8)))  float  v8f;

#define BQ 4
#define TQ 2048
#define VQ 2048
#define DQ 512
#define MALL (BQ * TQ)   // 8192

#define LDS_PITCH 40     // 80 B = 20 banks -> conflict-free b128 fragment reads

// ---------------------------------------------------------------------------
// Elementwise f32 -> bf16 convert
// ---------------------------------------------------------------------------
__global__ __launch_bounds__(256) void msq_cvt_bf16(const float* __restrict__ in,
                                                    __bf16* __restrict__ out,
                                                    unsigned long long n) {
  unsigned long long i = (unsigned long long)blockIdx.x * 256ull + threadIdx.x;
  if (i < n) out[i] = (__bf16)in[i];
}

// ---------------------------------------------------------------------------
// Transpose + convert:  in f32 [R,C] row-major  ->  out bf16 [C,R] row-major
// ---------------------------------------------------------------------------
__global__ __launch_bounds__(256) void msq_transpose_cvt(const float* __restrict__ in,
                                                         __bf16* __restrict__ out,
                                                         int R, int C) {
  unsigned long long i = (unsigned long long)blockIdx.x * 256ull + threadIdx.x;
  unsigned long long total = (unsigned long long)R * C;
  if (i < total) {
    int r = (int)(i / C);
    int c = (int)(i % C);
    out[(unsigned long long)c * R + r] = (__bf16)in[i];
  }
}

// ---------------------------------------------------------------------------
// gate = softmax(x @ Wg)  : one 128-thread block per row, K = V
// ---------------------------------------------------------------------------
__global__ __launch_bounds__(128) void msq_gate_softmax(const float* __restrict__ x,
                                                        const float* __restrict__ Wg,
                                                        float* __restrict__ gate,
                                                        int V) {
  __shared__ float red0[128], red1[128], red2[128];
  const int row = blockIdx.x;
  const int tid = threadIdx.x;
  const float* xr = x + (unsigned long long)row * V;
  float s0 = 0.f, s1 = 0.f, s2 = 0.f;
  for (int k = tid; k < V; k += 128) {
    float xv = xr[k];
    s0 += xv * Wg[k * 3 + 0];
    s1 += xv * Wg[k * 3 + 1];
    s2 += xv * Wg[k * 3 + 2];
  }
  red0[tid] = s0; red1[tid] = s1; red2[tid] = s2;
  __syncthreads();
  for (int off = 64; off > 0; off >>= 1) {
    if (tid < off) {
      red0[tid] += red0[tid + off];
      red1[tid] += red1[tid + off];
      red2[tid] += red2[tid + off];
    }
    __syncthreads();
  }
  if (tid == 0) {
    float a = red0[0], b = red1[0], c = red2[0];
    float m = fmaxf(a, fmaxf(b, c));
    float ea = __expf(a - m), eb = __expf(b - m), ec = __expf(c - m);
    float inv = 1.f / (ea + eb + ec);
    gate[row * 3 + 0] = ea * inv;
    gate[row * 3 + 1] = eb * inv;
    gate[row * 3 + 2] = ec * inv;
  }
}

// ---------------------------------------------------------------------------
// Generic WMMA GEMM:  C[M,N] = epilogue( A[M,K](bf16,rm) x Bt[N,K](bf16,rm)^T )
//
// Workgroup = 256 threads = 8 waves in a 2x4 grid; each wave computes a
// 32x32 block (2x2 WMMA 16x16 tiles) -> 64(M) x 128(N) macro tile, K step 32.
// Double-buffered LDS + register-staged global loads: loads for step s+1 are
// issued before the WMMAs of step s, so s_wait_loadcnt retires behind the
// compute phase instead of serializing with ds_store. 4 WMMAs per 8 ds_load
// (each A/B fragment reused twice). One barrier per K step.
// ---------------------------------------------------------------------------
template <bool OUT_BF16, bool STORE_T, bool ACC, bool DECAY, bool ROWSCALE>
__global__ __launch_bounds__(256) void msq_gemm_wmma(
    const __bf16* __restrict__ A, const __bf16* __restrict__ Bt,
    void* __restrict__ Cout, int M, int N, int K,
    const float* __restrict__ row_scale, int rs_stride,
    const float* __restrict__ decay_logit,
    const float* __restrict__ scale_ptr) {
  __shared__ __bf16 As[2][64 * LDS_PITCH];
  __shared__ __bf16 Bs[2][128 * LDS_PITCH];

  const int tid  = threadIdx.x;
  const int lane = tid & 31;
  const int wave = tid >> 5;
  const int wr   = wave >> 2;   // 0..1  (M direction)
  const int wc   = wave & 3;    // 0..3  (N direction)
  const int m0   = blockIdx.y * 64;
  const int n0   = blockIdx.x * 128;

  // cooperative staging coordinates (A: 64x32 = 4KB, B: 128x32 = 8KB)
  const int a_row = tid >> 2;          // 0..63
  const int a_col = (tid & 3) * 8;     // 0,8,16,24   (16B per thread)
  const int b_row = tid >> 1;          // 0..127
  const int b_col = (tid & 1) * 16;    // 0,16        (2x16B per thread)

  const __bf16* Aptr = A + (unsigned long long)(m0 + a_row) * K + a_col;
  const __bf16* Bptr = Bt + (unsigned long long)(n0 + b_row) * K + b_col;

  // preload K-step 0 through registers into buffer 0
  v8bf ra  = *(const v8bf*)(Aptr);
  v8bf rb0 = *(const v8bf*)(Bptr);
  v8bf rb1 = *(const v8bf*)(Bptr + 8);
  *(v8bf*)&As[0][a_row * LDS_PITCH + a_col]      = ra;
  *(v8bf*)&Bs[0][b_row * LDS_PITCH + b_col]      = rb0;
  *(v8bf*)&Bs[0][b_row * LDS_PITCH + b_col + 8]  = rb1;
  __syncthreads();

  v8f cc[2][2];
#pragma unroll
  for (int i = 0; i < 2; ++i)
#pragma unroll
    for (int j = 0; j < 2; ++j) cc[i][j] = (v8f){};

  const int kh  = lane >> 4;           // k-half select (0/1)
  const int l15 = lane & 15;
  const int amr = wr * 32 + l15;       // A row of fragment (i=0); +16 for i=1
  const int bnr = wc * 32 + l15;       // B col of fragment (j=0); +16 for j=1

  const int nsteps = K / 32;
  for (int s = 0; s < nsteps; ++s) {
    const int cur  = s & 1;
    const bool more = (s + 1) < nsteps;

    // issue global loads for the NEXT tile first (overlap with compute)
    if (more) {
      const int off = (s + 1) * 32;
      ra  = *(const v8bf*)(Aptr + off);
      rb0 = *(const v8bf*)(Bptr + off);
      rb1 = *(const v8bf*)(Bptr + off + 8);
    }
    if (s + 2 < nsteps) {   // warm L2/WGP$ two steps ahead
      __builtin_prefetch((const void*)(Aptr + (s + 2) * 32), 0, 1);
      __builtin_prefetch((const void*)(Bptr + (s + 2) * 32), 0, 1);
    }

    // build fragments from the CURRENT buffer
    v16bf af[2], bf2[2];
#pragma unroll
    for (int i = 0; i < 2; ++i) {
      const int row = (amr + i * 16) * LDS_PITCH;
      v8bf lo = *(const v8bf*)&As[cur][row + kh * 8];
      v8bf hi = *(const v8bf*)&As[cur][row + 16 + kh * 8];
#pragma unroll
      for (int e = 0; e < 8; ++e) { af[i][e] = lo[e]; af[i][e + 8] = hi[e]; }
    }
#pragma unroll
    for (int j = 0; j < 2; ++j) {
      const int row = (bnr + j * 16) * LDS_PITCH;
      v8bf lo = *(const v8bf*)&Bs[cur][row + kh * 8];
      v8bf hi = *(const v8bf*)&Bs[cur][row + 16 + kh * 8];
#pragma unroll
      for (int e = 0; e < 8; ++e) { bf2[j][e] = lo[e]; bf2[j][e + 8] = hi[e]; }
    }

#pragma unroll
    for (int i = 0; i < 2; ++i)
#pragma unroll
      for (int j = 0; j < 2; ++j)
        cc[i][j] = __builtin_amdgcn_wmma_f32_16x16x32_bf16(
            false, af[i], false, bf2[j], (short)0, cc[i][j], false, false);

    // stash the prefetched tile into the other buffer
    if (more) {
      const int nxt = cur ^ 1;
      *(v8bf*)&As[nxt][a_row * LDS_PITCH + a_col]     = ra;
      *(v8bf*)&Bs[nxt][b_row * LDS_PITCH + b_col]     = rb0;
      *(v8bf*)&Bs[nxt][b_row * LDS_PITCH + b_col + 8] = rb1;
      __syncthreads();
    }
  }

  // ---- epilogue ----
  const float scl = scale_ptr ? scale_ptr[0] : 1.0f;
  float logdecay = 0.f;
  if (DECAY) {
    const float lgt = decay_logit[0];
    const float dcy = 1.f / (1.f + __expf(-lgt));
    logdecay = __logf(dcy);
  }
#pragma unroll
  for (int i = 0; i < 2; ++i) {
#pragma unroll
    for (int j = 0; j < 2; ++j) {
      const int n = n0 + wc * 32 + j * 16 + l15;
#pragma unroll
      for (int r = 0; r < 8; ++r) {
        const int m = m0 + wr * 32 + i * 16 + kh * 8 + r;
        float val = cc[i][j][r] * scl;
        if (DECAY)    val = (n > m) ? val * __expf((float)(n - m - 1) * logdecay) : 0.f;
        if (ROWSCALE) val *= row_scale[(unsigned long long)m * rs_stride];
        const unsigned long long idx =
            STORE_T ? ((unsigned long long)n * M + m) : ((unsigned long long)m * N + n);
        if (OUT_BF16) {
          ((__bf16*)Cout)[idx] = (__bf16)val;
        } else {
          float* o = (float*)Cout;
          if (ACC) o[idx] += val; else o[idx] = val;
        }
      }
    }
  }
}

// ---------------------------------------------------------------------------
// Host orchestration
// ---------------------------------------------------------------------------
static inline unsigned long long msq_align(unsigned long long x) {
  return (x + 255ull) & ~255ull;
}

extern "C" void kernel_launch(void* const* d_in, const int* in_sizes, int n_in,
                              void* d_out, int out_size, void* d_ws, size_t ws_size,
                              hipStream_t stream) {
  (void)in_sizes; (void)n_in; (void)out_size; (void)ws_size;
  const float* x    = (const float*)d_in[0];   // [B,T,V]
  const float* Wq   = (const float*)d_in[1];   // [V,D]
  const float* Wk   = (const float*)d_in[2];   // [3,V,D]
  const float* Wv   = (const float*)d_in[3];   // [3,V,D]
  const float* Wo   = (const float*)d_in[4];   // [D,V]
  const float* Wg   = (const float*)d_in[5];   // [V,3]
  const float* dlog = (const float*)d_in[6];   // [3]
  const float* oscl = (const float*)d_in[7];   // scalar
  float* out = (float*)d_out;                  // [B,T,V] f32

  // ---- workspace carve-out ----
  char* w = (char*)d_ws;
  auto carve = [&](unsigned long long bytes) {
    void* p = (void*)w;
    w += msq_align(bytes);
    return p;
  };
  __bf16* xb   = (__bf16*)carve((unsigned long long)MALL * VQ * 2);       // x bf16
  __bf16* Wqt  = (__bf16*)carve((unsigned long long)DQ * VQ * 2);         // [D,V]
  __bf16* Wkt  = (__bf16*)carve(3ull * DQ * VQ * 2);                      // 3x[D,V]
  __bf16* Wvt  = (__bf16*)carve(3ull * DQ * VQ * 2);                      // 3x[D,V]
  __bf16* Wot  = (__bf16*)carve((unsigned long long)VQ * DQ * 2);         // [V,D]
  __bf16* qb   = (__bf16*)carve((unsigned long long)MALL * DQ * 2);       // [B*T,D]
  __bf16* kb   = (__bf16*)carve(3ull * MALL * DQ * 2);                    // 3x[B*T,D]
  __bf16* vTb  = (__bf16*)carve(3ull * BQ * DQ * TQ * 2);                 // 3x B x [D,T]
  __bf16* scb  = (__bf16*)carve((unsigned long long)TQ * TQ * 2);         // [T,T] reused
  float*  blend= (float*) carve((unsigned long long)MALL * DQ * 4);       // [B*T,D] f32
  __bf16* blb  = (__bf16*)carve((unsigned long long)MALL * DQ * 2);       // [B*T,D] bf16
  float*  gate = (float*) carve((unsigned long long)MALL * 3 * 4);        // [B*T,3]

  // 1) x -> bf16
  {
    unsigned long long n = (unsigned long long)MALL * VQ;
    msq_cvt_bf16<<<(unsigned)((n + 255) / 256), 256, 0, stream>>>(x, xb, n);
  }
  // 2) weight transpose+convert:  W[K,N] f32 -> Wt[N,K] bf16
  {
    unsigned grid = (unsigned)(((unsigned long long)VQ * DQ + 255) / 256);
    msq_transpose_cvt<<<grid, 256, 0, stream>>>(Wq, Wqt, VQ, DQ);
    for (int i = 0; i < 3; ++i) {
      msq_transpose_cvt<<<grid, 256, 0, stream>>>(Wk + (unsigned long long)i * VQ * DQ,
                                                  Wkt + (unsigned long long)i * DQ * VQ, VQ, DQ);
      msq_transpose_cvt<<<grid, 256, 0, stream>>>(Wv + (unsigned long long)i * VQ * DQ,
                                                  Wvt + (unsigned long long)i * DQ * VQ, VQ, DQ);
    }
    msq_transpose_cvt<<<grid, 256, 0, stream>>>(Wo, Wot, DQ, VQ);
  }
  // 3) gate = softmax(x @ Wg)
  msq_gate_softmax<<<MALL, 128, 0, stream>>>(x, Wg, gate, VQ);

  // 4) q = xb @ Wq   (M=8192, N=512, K=2048), bf16 out
  {
    dim3 grid(DQ / 128, MALL / 64);
    msq_gemm_wmma<true, false, false, false, false><<<grid, 256, 0, stream>>>(
        xb, Wqt, qb, MALL, DQ, VQ, nullptr, 0, nullptr, nullptr);
  }
  // 5) k_i = xb @ Wk[i]  (bf16 out, normal layout — already the Bt layout for q@k^T)
  for (int i = 0; i < 3; ++i) {
    dim3 grid(DQ / 128, MALL / 64);
    msq_gemm_wmma<true, false, false, false, false><<<grid, 256, 0, stream>>>(
        xb, Wkt + (unsigned long long)i * DQ * VQ,
        kb + (unsigned long long)i * MALL * DQ,
        MALL, DQ, VQ, nullptr, 0, nullptr, nullptr);
  }
  // 6) vT_{i,b} = (x_b @ Wv[i])^T  (per-batch, transposed store -> [D,T])
  for (int i = 0; i < 3; ++i)
    for (int b = 0; b < BQ; ++b) {
      dim3 grid(DQ / 128, TQ / 64);
      msq_gemm_wmma<true, true, false, false, false><<<grid, 256, 0, stream>>>(
          xb + (unsigned long long)b * TQ * VQ,
          Wvt + (unsigned long long)i * DQ * VQ,
          vTb + ((unsigned long long)i * BQ + b) * DQ * TQ,
          TQ, DQ, VQ, nullptr, 0, nullptr, nullptr);
    }
  // 7) per (b, i): masked scores then gated accumulate into blend
  for (int b = 0; b < BQ; ++b) {
    for (int i = 0; i < 3; ++i) {
      // scores = (q_b k_{i,b}^T) * w   -> scb bf16 [T,T]
      {
        dim3 grid(TQ / 128, TQ / 64);
        msq_gemm_wmma<true, false, false, true, false><<<grid, 256, 0, stream>>>(
            qb + (unsigned long long)b * TQ * DQ,
            kb + ((unsigned long long)i * MALL + (unsigned long long)b * TQ) * DQ,
            scb, TQ, TQ, DQ, nullptr, 0, dlog + i, nullptr);
      }
      // blend_b (+)= gate[:,i] * (scores @ v_{i,b})   (f32, row-scaled)
      {
        dim3 grid(DQ / 128, TQ / 64);
        const float* rs = gate + (unsigned long long)b * TQ * 3 + i;
        float* ob = blend + (unsigned long long)b * TQ * DQ;
        if (i == 0)
          msq_gemm_wmma<false, false, false, false, true><<<grid, 256, 0, stream>>>(
              scb, vTb + ((unsigned long long)i * BQ + b) * DQ * TQ,
              ob, TQ, DQ, TQ, rs, 3, nullptr, nullptr);
        else
          msq_gemm_wmma<false, false, true, false, true><<<grid, 256, 0, stream>>>(
              scb, vTb + ((unsigned long long)i * BQ + b) * DQ * TQ,
              ob, TQ, DQ, TQ, rs, 3, nullptr, nullptr);
      }
    }
  }
  // 8) blend -> bf16
  {
    unsigned long long n = (unsigned long long)MALL * DQ;
    msq_cvt_bf16<<<(unsigned)((n + 255) / 256), 256, 0, stream>>>(blend, blb, n);
  }
  // 9) out = (blend @ Wo) * out_scale   (M=8192, N=2048, K=512), f32 out
  {
    dim3 grid(VQ / 128, MALL / 64);
    msq_gemm_wmma<false, false, false, false, false><<<grid, 256, 0, stream>>>(
        blb, Wot, out, MALL, VQ, DQ, nullptr, 0, nullptr, oscl);
  }
}
